// SmallSNN_58067957842529
// MI455X (gfx1250) — matrix operand order repaired and checked by reading, not compiled, not run on hardware
//
#include <hip/hip_runtime.h>
#include <hip/hip_bf16.h>
#include <stdint.h>

// ---------------------------------------------------------------------------
// SNN forward: layer1 GEMM (bf16 WMMA, fp32 accum) + fused leaky-IF recurrence
// ---------------------------------------------------------------------------

typedef __attribute__((ext_vector_type(16))) __bf16          v16bf;
typedef __attribute__((ext_vector_type(16))) unsigned short  v16u;
typedef __attribute__((ext_vector_type(8)))  unsigned short  u16x8;
typedef __attribute__((ext_vector_type(4)))  unsigned short  u16x4;
typedef __attribute__((ext_vector_type(8)))  float           v8f;

#define T_STEPS   50
#define BATCH     128
#define NH        256
#define NO        11
#define KDIM      16384
#define MROWS     (T_STEPS * BATCH)   // 6400
#define M_TILE    32
#define K_TILE    32
#define K_SLABS   (KDIM / K_TILE)     // 512
#define LDS_STRIDE 40                 // 32 + pad (80B rows -> conflict-free)

__device__ __forceinline__ unsigned short f32_bf16(float f) {
  unsigned int u = __builtin_bit_cast(unsigned int, f);
  u += 0x7FFFu + ((u >> 16) & 1u);          // round-to-nearest-even
  return (unsigned short)(u >> 16);
}

// ---------------------------------------------------------------------------
// Kernel 0: W1 fp32 -> bf16 (row-major [NH][KDIM]), done once per launch
// ---------------------------------------------------------------------------
__global__ void __launch_bounds__(256)
w1_to_bf16(const float* __restrict__ w1, unsigned short* __restrict__ w1bf) {
  int i = (blockIdx.x * 256 + threadIdx.x) * 4;
  float4 v = *(const float4*)(w1 + i);
  u16x4 o;
  o[0] = f32_bf16(v.x); o[1] = f32_bf16(v.y);
  o[2] = f32_bf16(v.z); o[3] = f32_bf16(v.w);
  *(u16x4*)(w1bf + i) = o;
}

// ---------------------------------------------------------------------------
// Kernel 1: cur1[m][n] = sum_k x[m][k]*W1[n][k] + b1[n]
//   m = t*BATCH + b  (all 6400 rows batched; recurrence applied later)
//   Block: 32 M-rows x 256 N (8 waves, each wave: 2 M-subtiles x 2 N-subtiles)
//   x staged fp32->bf16 into double-buffered LDS; W1 read as bf16 from L2.
// ---------------------------------------------------------------------------
__global__ void __launch_bounds__(256, 2)
snn_gemm1(const float* __restrict__ x, const unsigned short* __restrict__ w1bf,
          const float* __restrict__ b1, float* __restrict__ cur1) {
  __shared__ __align__(16) unsigned short xs[2][M_TILE][LDS_STRIDE];

  const int tid  = threadIdx.x;
  const int m0   = blockIdx.x * M_TILE;
  const int wv   = tid >> 5;        // wave 0..7 -> N strip of 32
  const int lane = tid & 31;
  const int l16  = lane & 15;
  const int hi   = lane >> 4;

  // cooperative x-tile loader coords: 8 threads/row, float4 each
  const int lr = tid >> 3;          // row 0..31
  const int lk = (tid & 7) << 2;    // k 0..28 step 4

  v8f c[2][2] = {};

  // prologue: slab 0 -> buffer 0
  {
    float4 v = *(const float4*)(x + (size_t)(m0 + lr) * KDIM + lk);
    u16x4 o;
    o[0] = f32_bf16(v.x); o[1] = f32_bf16(v.y);
    o[2] = f32_bf16(v.z); o[3] = f32_bf16(v.w);
    *(u16x4*)&xs[0][lr][lk] = o;
  }
  __syncthreads();

  for (int ks = 0; ks < K_SLABS; ++ks) {
    const int bi = ks & 1;

    // prefetch next slab into the other buffer (overlaps with WMMA below)
    if (ks + 1 < K_SLABS) {
      float4 v = *(const float4*)(x + (size_t)(m0 + lr) * KDIM + (ks + 1) * K_TILE + lk);
      u16x4 o;
      o[0] = f32_bf16(v.x); o[1] = f32_bf16(v.y);
      o[2] = f32_bf16(v.z); o[3] = f32_bf16(v.w);
      *(u16x4*)&xs[bi ^ 1][lr][lk] = o;
    }

    // A fragments (16x32 bf16, contiguous k order used here and in B: lanes 0-15 K{0..15}, lanes 16-31 K{16..31})
    v16bf a[2];
#pragma unroll
    for (int mi = 0; mi < 2; ++mi) {
      union { u16x8 h[2]; v16u v; } au;
      const unsigned short* ap = &xs[bi][mi * 16 + l16][hi * 16];
      au.h[0] = *(const u16x8*)ap;          // K = 16*hi + 0..7
      au.h[1] = *(const u16x8*)(ap + 8);    // K = 16*hi + 8..15
      a[mi] = __builtin_bit_cast(v16bf, au.v);
    }

    // B fragments (32x16: lane n, 16 contiguous K = 16*hi..16*hi+15 of W1 row n)
#pragma unroll
    for (int ni = 0; ni < 2; ++ni) {
      const int col = wv * 32 + ni * 16 + l16;
      v16u bu = *(const v16u*)(w1bf + (size_t)col * KDIM + ks * K_TILE + hi * 16);
      v16bf b = __builtin_bit_cast(v16bf, bu);
#pragma unroll
      for (int mi = 0; mi < 2; ++mi)
        c[mi][ni] = __builtin_amdgcn_wmma_f32_16x16x32_bf16(
            false, a[mi], false, b, (short)0, c[mi][ni], false, false);
    }
    __syncthreads();
  }

  // epilogue: C/D layout -> VGPR r holds M = 8*hi + r, N = lane&15
#pragma unroll
  for (int ni = 0; ni < 2; ++ni) {
    const int n = wv * 32 + ni * 16 + l16;
    const float bias = b1[n];
#pragma unroll
    for (int mi = 0; mi < 2; ++mi) {
#pragma unroll
      for (int r = 0; r < 8; ++r) {
        const int m = m0 + mi * 16 + hi * 8 + r;
        cur1[(size_t)m * NH + n] = c[mi][ni][r] + bias;
      }
    }
  }
}

// ---------------------------------------------------------------------------
// Kernel 2: fused leaky-IF recurrence (layer 1) + layer 2 + recurrence
//   One block per batch element; thread h owns hidden unit h's mem1.
//   spk1 never touches global memory: wave shfl-reduction + ds_add_f32.
// ---------------------------------------------------------------------------
__global__ void __launch_bounds__(256)
snn_recur(const float* __restrict__ cur1, const float* __restrict__ w2,
          const float* __restrict__ b2, float* __restrict__ out) {
  const int b    = blockIdx.x;
  const int h    = threadIdx.x;     // hidden unit 0..255
  const int lane = h & 31;

  __shared__ float acc[NO];

  float w2c[NO];
#pragma unroll
  for (int o = 0; o < NO; ++o) w2c[o] = w2[o * NH + h];

  float mem1 = 0.f;
  float mem2 = 0.f;
  const float bias2 = (h < NO) ? b2[h] : 0.f;

  for (int t = 0; t < T_STEPS; ++t) {
    if (h < NO) acc[h] = 0.f;
    __syncthreads();

    // layer-1 leaky integrate & fire, subtract-reset
    const float cur = cur1[((size_t)t * BATCH + b) * NH + h];
    mem1 = 0.9f * mem1 + cur;
    const float s = (mem1 - 1.0f > 0.f) ? 1.f : 0.f;
    mem1 -= s;

    // layer-2: sum_h s[h] * W2[o][h]  (wave reduce + LDS float atomic)
#pragma unroll
    for (int o = 0; o < NO; ++o) {
      float p = s * w2c[o];
      p += __shfl_xor(p, 16);
      p += __shfl_xor(p, 8);
      p += __shfl_xor(p, 4);
      p += __shfl_xor(p, 2);
      p += __shfl_xor(p, 1);
      if (lane == 0) atomicAdd(&acc[o], p);
    }
    __syncthreads();

    if (h < NO) {
      const float c2 = acc[h] + bias2;
      mem2 = 0.9f * mem2 + c2;
      const float s2 = (mem2 - 1.0f > 0.f) ? 1.f : 0.f;
      mem2 -= s2;
      out[((size_t)t * BATCH + b) * NO + h] = s2;
    }
    // next iteration's zeroing is done by the same threads that just read acc,
    // and the top barrier orders it before any new atomicAdd -> race-free.
  }
}

// ---------------------------------------------------------------------------
extern "C" void kernel_launch(void* const* d_in, const int* in_sizes, int n_in,
                              void* d_out, int out_size, void* d_ws, size_t ws_size,
                              hipStream_t stream) {
  const float* x  = (const float*)d_in[0];   // [50,128,128,128]
  const float* W1 = (const float*)d_in[1];   // [256,16384]
  const float* b1 = (const float*)d_in[2];   // [256]
  const float* W2 = (const float*)d_in[3];   // [11,256]
  const float* b2 = (const float*)d_in[4];   // [11]
  float* out = (float*)d_out;                // [50,128,11]

  // workspace layout
  unsigned short* w1bf = (unsigned short*)d_ws;                       // 8.39 MB
  float* cur1 = (float*)((char*)d_ws + (size_t)NH * KDIM * sizeof(unsigned short)); // 6.55 MB

  // 1) convert W1 once
  w1_to_bf16<<<(NH * KDIM / 4) / 256, 256, 0, stream>>>(W1, w1bf);

  // 2) batched layer-1 GEMM over all (t,b) rows
  snn_gemm1<<<MROWS / M_TILE, 256, 0, stream>>>(x, w1bf, b1, cur1);

  // 3) fused recurrences + layer 2
  snn_recur<<<BATCH, 256, 0, stream>>>(cur1, W2, b2, out);
}
